// QuantumFourierAttention_89215060672623
// MI455X (gfx1250) — compile-verified, matrix operand order checked by reading, not run on hardware
//
#include <hip/hip_runtime.h>
#include <math.h>

// ---------------------------------------------------------------------------
// QuantumFourierAttention for MI455X (gfx1250, wave32, WMMA).
// All GEMMs use v_wmma_f32_16x16x32_f16 (f16 in, f32 accumulate).
// Workspace layout (bytes), total 112 MB:
//   [  0MB, 16MB) Cr   fp32 [4096x1024]   projection real / attn outF real
//   [ 16MB, 32MB) Ci   fp32 [4096x1024]   projection imag / attn outF imag
//   [ 32MB, 48MB) Hr   fp32 [B,H,N,64]    LN'd real (head layout) / ifft real
//   [ 48MB, 64MB) Hi   fp32 [B,H,N,64]    LN'd imag (head layout) / ifft imag
//   [ 64MB, 80MB) Qf   f16  [B,H,N,128]   FFT(Q) [re|im] cat
//   [ 80MB, 96MB) Kf   f16  [B,H,N,128]
//   [ 96MB,112MB) Vf   f16  [B,H,N,128]
// ---------------------------------------------------------------------------

typedef __attribute__((ext_vector_type(16))) _Float16 v16h;
typedef __attribute__((ext_vector_type(8)))  _Float16 v8h;
typedef __attribute__((ext_vector_type(8)))  float    v8f;

#define WMMA_F16(a, b, c) \
  __builtin_amdgcn_wmma_f32_16x16x32_f16(false, (a), false, (b), (short)0, (c), false, false)

// Build one 16x32 f16 A/B fragment from an LDS tile (row-major, strideH in
// halves).  CDNA5 layout: lanes 0-15 hold K = {0..7, 16..23}, lanes 16-31 hold
// K = {8..15, 24..31}; M/N = lane%16.  Two 16B LDS loads per fragment.
__device__ __forceinline__ v16h frag_ld(const _Float16* base, int strideH,
                                        int row, int koff, int lane) {
  const int lh = lane & 15;
  const int b8 = (lane & 16) ? 8 : 0;
  const _Float16* p = base + (size_t)(row + lh) * strideH + (koff + b8);
  v8h lo = *(const v8h*)(p);
  v8h hi = *(const v8h*)(p + 16);
  return __builtin_shufflevector(lo, hi, 0,1,2,3,4,5,6,7,8,9,10,11,12,13,14,15);
}

// ---------------------------------------------------------------------------
// Complex GEMM as one real GEMM with K=2048:
//   C[t][n] = sum_k [X0|X1][t][k] * [W0 | w1sgn*W1][n][k]  + bias[n]
// a_mode 0: X* are row-major [4096 x 1024]
// a_mode 1: X* are head-layout [B,H,N,64]; k -> (h = k/64, d = k%64)
// fp32 loaded from global, converted to f16 while staging into LDS.
// Block tile 128x128, double-buffered LDS, 8 waves; each wave owns a 32x64
// sub-tile -> 8 WMMAs per K-step from 6 fragment loads (A/B frag reuse).
// ---------------------------------------------------------------------------
__global__ __launch_bounds__(256)
void gemm_cplx_wmma(const float* __restrict__ X0, const float* __restrict__ X1,
                    int a_mode,
                    const float* __restrict__ W0, const float* __restrict__ W1,
                    float w1sgn, const float* __restrict__ bias,
                    float* __restrict__ C, int ldc) {
  __shared__ __attribute__((aligned(16))) _Float16 As[2][128][40];  // 128x32+pad
  __shared__ __attribute__((aligned(16))) _Float16 Bs[2][128][40];

  const int tid  = threadIdx.x;
  const int lane = tid & 31;
  const int w    = tid >> 5;      // 8 waves (wave32)
  const int wm   = w >> 1;        // 0..3 : 32-row stripe
  const int wn   = w & 1;         // 0..1 : 64-col stripe
  const int bn0  = blockIdx.x * 128;
  const int bm0  = blockIdx.y * 128;
  const int lr   = tid >> 1;           // 0..127 tile row to stage
  const int lc   = (tid & 1) * 16;     // 0 or 16 (16 contiguous halves)

  v8f acc[2][4];
  for (int i = 0; i < 2; ++i)
    for (int j = 0; j < 4; ++j)
      acc[i][j] = (v8f){0.f,0.f,0.f,0.f,0.f,0.f,0.f,0.f};

  // stage one 128x32 A (or B) tile slice for K-step kt into buffer `buf`
  auto stageA = [&](int kt, int buf) {
    const int k0    = kt * 32;
    const int phase = (k0 >= 1024);
    const int kloc  = (k0 - (phase ? 1024 : 0)) + lc;
    const float* Xp = phase ? X1 : X0;
    const int t = bm0 + lr;
    const float* p;
    if (a_mode == 0) {
      p = Xp + (size_t)t * 1024 + kloc;
    } else {
      const int b = t >> 10, n = t & 1023;
      p = Xp + (((size_t)(b * 16 + (kloc >> 6)) * 1024) + n) * 64 + (kloc & 63);
    }
    float4 f0 = *(const float4*)(p + 0);
    float4 f1 = *(const float4*)(p + 4);
    float4 f2 = *(const float4*)(p + 8);
    float4 f3 = *(const float4*)(p + 12);
    __builtin_prefetch(p + 32, 0, 1);    // speculative: K-step after next
    v8h h0, h1;
    h0[0]=(_Float16)f0.x; h0[1]=(_Float16)f0.y; h0[2]=(_Float16)f0.z; h0[3]=(_Float16)f0.w;
    h0[4]=(_Float16)f1.x; h0[5]=(_Float16)f1.y; h0[6]=(_Float16)f1.z; h0[7]=(_Float16)f1.w;
    h1[0]=(_Float16)f2.x; h1[1]=(_Float16)f2.y; h1[2]=(_Float16)f2.z; h1[3]=(_Float16)f2.w;
    h1[4]=(_Float16)f3.x; h1[5]=(_Float16)f3.y; h1[6]=(_Float16)f3.z; h1[7]=(_Float16)f3.w;
    *(v8h*)&As[buf][lr][lc]     = h0;
    *(v8h*)&As[buf][lr][lc + 8] = h1;
  };
  auto stageB = [&](int kt, int buf) {
    const int k0    = kt * 32;
    const int phase = (k0 >= 1024);
    const int kloc  = (k0 - (phase ? 1024 : 0)) + lc;
    const float* Wp = phase ? W1 : W0;
    const float s   = phase ? w1sgn : 1.f;
    const float* p  = Wp + (size_t)(bn0 + lr) * 1024 + kloc;
    float4 f0 = *(const float4*)(p + 0);
    float4 f1 = *(const float4*)(p + 4);
    float4 f2 = *(const float4*)(p + 8);
    float4 f3 = *(const float4*)(p + 12);
    v8h h0, h1;
    h0[0]=(_Float16)(s*f0.x); h0[1]=(_Float16)(s*f0.y); h0[2]=(_Float16)(s*f0.z); h0[3]=(_Float16)(s*f0.w);
    h0[4]=(_Float16)(s*f1.x); h0[5]=(_Float16)(s*f1.y); h0[6]=(_Float16)(s*f1.z); h0[7]=(_Float16)(s*f1.w);
    h1[0]=(_Float16)(s*f2.x); h1[1]=(_Float16)(s*f2.y); h1[2]=(_Float16)(s*f2.z); h1[3]=(_Float16)(s*f2.w);
    h1[4]=(_Float16)(s*f3.x); h1[5]=(_Float16)(s*f3.y); h1[6]=(_Float16)(s*f3.z); h1[7]=(_Float16)(s*f3.w);
    *(v8h*)&Bs[buf][lr][lc]     = h0;
    *(v8h*)&Bs[buf][lr][lc + 8] = h1;
  };

  stageA(0, 0);
  stageB(0, 0);

  for (int kt = 0; kt < 64; ++kt) {           // K = 2048, BK = 32
    __syncthreads();                          // buf[kt&1] ready; buf^1 free
    const int buf = kt & 1;
    if (kt + 1 < 64) {                        // overlap HBM with WMMA
      stageA(kt + 1, buf ^ 1);
      stageB(kt + 1, buf ^ 1);
    }
    v16h a0 = frag_ld(&As[buf][0][0], 40, wm * 32,      0, lane);
    v16h a1 = frag_ld(&As[buf][0][0], 40, wm * 32 + 16, 0, lane);
    for (int j = 0; j < 4; ++j) {
      v16h bj = frag_ld(&Bs[buf][0][0], 40, wn * 64 + j * 16, 0, lane);
      acc[0][j] = WMMA_F16(a0, bj, acc[0][j]);
      acc[1][j] = WMMA_F16(a1, bj, acc[1][j]);
    }
  }

  // epilogue: C layout -> lanes 0-15 rows r, lanes 16-31 rows 8+r
  const int hi8 = (lane & 16) ? 8 : 0;
  const int lh  = lane & 15;
  for (int j = 0; j < 4; ++j) {
    const int col = bn0 + wn * 64 + j * 16 + lh;
    const float bj = bias[col];
    for (int i = 0; i < 2; ++i) {
      const int row0 = bm0 + wm * 32 + i * 16 + hi8;
      for (int r = 0; r < 8; ++r)
        C[(size_t)(row0 + r) * ldc + col] = acc[i][j][r] + bj;
    }
  }
}

// ---------------------------------------------------------------------------
// Complex LayerNorm on magnitudes, phase preserved; fuses the head transpose:
// input row-major [4096 x 1024] -> output [B,H,N,64].
// ---------------------------------------------------------------------------
__global__ __launch_bounds__(256)
void cln_kernel(const float* __restrict__ Cr, const float* __restrict__ Ci,
                const float* __restrict__ g, const float* __restrict__ be,
                float* __restrict__ Hr, float* __restrict__ Hi) {
  __shared__ float rs[256], rq[256];
  const int t   = blockIdx.x;          // token 0..4095
  const int tid = threadIdx.x;
  const int b   = t >> 10, n = t & 1023;

  float mr[4], mi[4], mg[4];
  float s = 0.f, q = 0.f;
  for (int k = 0; k < 4; ++k) {
    const int j = tid + k * 256;
    const float r = Cr[(size_t)t * 1024 + j];
    const float i = Ci[(size_t)t * 1024 + j];
    const float m = sqrtf(r * r + i * i);
    mr[k] = r; mi[k] = i; mg[k] = m;
    s += m; q += m * m;
  }
  rs[tid] = s; rq[tid] = q;
  __syncthreads();
  for (int off = 128; off > 0; off >>= 1) {
    if (tid < off) { rs[tid] += rs[tid + off]; rq[tid] += rq[tid + off]; }
    __syncthreads();
  }
  const float mean = rs[0] * (1.f / 1024.f);
  const float var  = rq[0] * (1.f / 1024.f) - mean * mean;
  const float rstd = rsqrtf(var + 1e-6f);

  for (int k = 0; k < 4; ++k) {
    const int j  = tid + k * 256;
    const float nm  = (mg[k] - mean) * rstd * g[j] + be[j];
    const bool  nz  = mg[k] > 0.f;
    const float inv = nz ? 1.f / mg[k] : 0.f;
    const float cr  = nz ? mr[k] * inv : 1.f;   // cos(angle(0)) == 1
    const float ci  = nz ? mi[k] * inv : 0.f;
    const int h = j >> 6, d = j & 63;
    const size_t o = ((size_t)(b * 16 + h) * 1024 + n) * 64 + d;
    Hr[o] = nm * cr;
    Hi[o] = nm * ci;
  }
}

// ---------------------------------------------------------------------------
// 1024-point radix-2 DIT FFT along N, 4 channels per workgroup, fp32 in LDS.
// sign = -1 forward, +1 inverse; ortho norm (1/32 both ways).
// If outCat != null, writes f16 [B,H,N,128] as [re(0..63) | im(64..127)].
// ---------------------------------------------------------------------------
__global__ __launch_bounds__(256)
void fft1024_kernel(const float* __restrict__ inR, const float* __restrict__ inI,
                    float sign, float* __restrict__ outR, float* __restrict__ outI,
                    _Float16* __restrict__ outCat) {
  __shared__ float Sr[1024][5];   // 4 channels + pad
  __shared__ float Si[1024][5];
  const int tid = threadIdx.x;
  const int bh  = blockIdx.x >> 4;          // (b*16+h)
  const int d0  = (blockIdx.x & 15) * 4;    // channel chunk
  const size_t base = (size_t)bh * 1024 * 64;

  // load in bit-reversed order (DIT)
  for (int v = tid; v < 4096; v += 256) {
    const int n = v >> 2, dd = v & 3;
    const int rn = (int)(__brev((unsigned)n) >> 22);
    Sr[n][dd] = inR[base + (size_t)rn * 64 + d0 + dd];
    Si[n][dd] = inI[base + (size_t)rn * 64 + d0 + dd];
  }
  for (int s = 1; s <= 10; ++s) {
    __syncthreads();
    const int half = 1 << (s - 1);
    for (int it = tid; it < 2048; it += 256) {
      const int bf = it >> 2, dd = it & 3;
      const int pos = bf & (half - 1);
      const int grp = bf >> (s - 1);
      const int i0 = (grp << s) + pos, i1 = i0 + half;
      const float ang = sign * 6.28318530717958647692f * (float)pos / (float)(1 << s);
      float sn, cs;
      __sincosf(ang, &sn, &cs);
      const float r1 = Sr[i1][dd], q1 = Si[i1][dd];
      const float tr = cs * r1 - sn * q1;
      const float ti = cs * q1 + sn * r1;
      const float r0 = Sr[i0][dd], q0 = Si[i0][dd];
      Sr[i1][dd] = r0 - tr; Si[i1][dd] = q0 - ti;
      Sr[i0][dd] = r0 + tr; Si[i0][dd] = q0 + ti;
    }
  }
  __syncthreads();
  const float nrm = 0.03125f;   // 1/sqrt(1024)
  for (int v = tid; v < 4096; v += 256) {
    const int n = v >> 2, dd = v & 3;
    const float r = Sr[n][dd] * nrm, q = Si[n][dd] * nrm;
    if (outCat) {
      const size_t o = ((size_t)bh * 1024 + n) * 128 + d0 + dd;
      outCat[o]      = (_Float16)r;
      outCat[o + 64] = (_Float16)q;
    } else {
      const size_t o = base + (size_t)n * 64 + d0 + dd;
      outR[o] = r; outI[o] = q;
    }
  }
}

// ---------------------------------------------------------------------------
// Flash attention on Fourier coefficients.  One workgroup = one (b,h) and a
// 64-row query block.  scores.real = Qcat @ Kcat^T (K=128); out = P @ Vcat
// (N=128, columns 0..63 real, 64..127 imag).  Online softmax, fp32 stats.
// ---------------------------------------------------------------------------
__global__ __launch_bounds__(256)
void flash_attn_wmma(const _Float16* __restrict__ Qf, const _Float16* __restrict__ Kf,
                     const _Float16* __restrict__ Vf,
                     float* __restrict__ OFr, float* __restrict__ OFi) {
  __shared__ __attribute__((aligned(16))) _Float16 Ks[64][136];   // K tile (also stages Q)
  __shared__ __attribute__((aligned(16))) _Float16 VsT[128][72];  // V transposed [d][kv]
  __shared__ float                                 Ss[64][68];    // scores fp32
  __shared__ __attribute__((aligned(16))) _Float16 Ps[64][72];    // exp probs f16
  __shared__ float mI[64], lI[64], alphaS[64];

  const int tid  = threadIdx.x;
  const int lane = tid & 31;
  const int w    = tid >> 5;
  const int wm   = w & 3;          // 16-row stripe
  const int wn   = w >> 2;         // col half
  const int bh   = blockIdx.x >> 4;
  const int qb   = blockIdx.x & 15;
  const int lh   = lane & 15;
  const int hi8  = (lane & 16) ? 8 : 0;

  // stage Q block into Ks, build persistent A fragments (4 x K32)
  const size_t qbase = ((size_t)bh * 1024 + qb * 64) * 128;
  for (int v = tid; v < 64 * 16; v += 256) {
    const int r = v >> 4, c8 = (v & 15) * 8;
    *(v8h*)&Ks[r][c8] = *(const v8h*)(Qf + qbase + (size_t)r * 128 + c8);
  }
  if (tid < 64) { mI[tid] = -1e30f; lI[tid] = 0.f; }
  __syncthreads();
  v16h qf[4];
  for (int kk = 0; kk < 4; ++kk) qf[kk] = frag_ld(&Ks[0][0], 136, wm * 16, kk * 32, lane);
  __syncthreads();

  v8f oacc[4];
  for (int t4 = 0; t4 < 4; ++t4) oacc[t4] = (v8f){0.f,0.f,0.f,0.f,0.f,0.f,0.f,0.f};

  for (int kb = 0; kb < 16; ++kb) {
    const size_t kvbase = ((size_t)bh * 1024 + kb * 64) * 128;
    for (int v = tid; v < 64 * 16; v += 256) {
      const int r = v >> 4, c8 = (v & 15) * 8;
      *(v8h*)&Ks[r][c8] = *(const v8h*)(Kf + kvbase + (size_t)r * 128 + c8);
      v8h vv = *(const v8h*)(Vf + kvbase + (size_t)r * 128 + c8);
      for (int j = 0; j < 8; ++j) VsT[c8 + j][r] = vv[j];
    }
    __syncthreads();

    // S = Q @ K^T * scale  (each wave: two 16x16 tiles, K=128)
    for (int t = 0; t < 2; ++t) {
      v8f sacc = {0.f,0.f,0.f,0.f,0.f,0.f,0.f,0.f};
      const int n0s = wn * 32 + t * 16;
      for (int kk = 0; kk < 4; ++kk) {
        v16h bf = frag_ld(&Ks[0][0], 136, n0s, kk * 32, lane);
        sacc = WMMA_F16(qf[kk], bf, sacc);
      }
      for (int r = 0; r < 8; ++r)
        Ss[wm * 16 + hi8 + r][n0s + lh] = sacc[r] * 0.125f;  // 1/sqrt(64)
    }
    __syncthreads();

    // online softmax stats, one row per thread
    if (tid < 64) {
      const float mold = mI[tid];
      float bmax = -1e30f;
      for (int c = 0; c < 64; ++c) bmax = fmaxf(bmax, Ss[tid][c]);
      const float mnew = fmaxf(mold, bmax);
      float sum = 0.f;
      for (int c = 0; c < 64; ++c) {
        const float e = __expf(Ss[tid][c] - mnew);
        Ps[tid][c] = (_Float16)e;
        sum += e;
      }
      const float al = __expf(mold - mnew);
      lI[tid] = lI[tid] * al + sum;
      mI[tid] = mnew;
      alphaS[tid] = al;
    }
    __syncthreads();

    // rescale O and accumulate P @ Vcat
    float al[8];
    for (int r = 0; r < 8; ++r) al[r] = alphaS[wm * 16 + hi8 + r];
    for (int t4 = 0; t4 < 4; ++t4)
      for (int r = 0; r < 8; ++r) oacc[t4][r] *= al[r];

    v16h pf0 = frag_ld(&Ps[0][0], 72, wm * 16, 0,  lane);
    v16h pf1 = frag_ld(&Ps[0][0], 72, wm * 16, 32, lane);
    for (int t4 = 0; t4 < 4; ++t4) {
      const int d0 = wn * 64 + t4 * 16;
      v16h vf0 = frag_ld(&VsT[0][0], 72, d0, 0,  lane);
      v16h vf1 = frag_ld(&VsT[0][0], 72, d0, 32, lane);
      oacc[t4] = WMMA_F16(pf0, vf0, oacc[t4]);
      oacc[t4] = WMMA_F16(pf1, vf1, oacc[t4]);
    }
    __syncthreads();
  }

  // epilogue: divide by l, split re/im halves, write fp32 [B,H,N,64]
  for (int t4 = 0; t4 < 4; ++t4) {
    const int col = wn * 64 + t4 * 16 + lh;
    for (int r = 0; r < 8; ++r) {
      const int row = wm * 16 + hi8 + r;
      const float o = oacc[t4][r] / lI[row];
      const size_t gi = ((size_t)bh * 1024 + qb * 64 + row) * 64;
      if (col < 64) OFr[gi + col] = o;
      else          OFi[gi + col - 64] = o;
    }
  }
}

// ---------------------------------------------------------------------------
extern "C" void kernel_launch(void* const* d_in, const int* in_sizes, int n_in,
                              void* d_out, int out_size, void* d_ws, size_t ws_size,
                              hipStream_t stream) {
  (void)in_sizes; (void)n_in; (void)out_size; (void)ws_size;

  const float* xr = (const float*)d_in[0];
  const float* xi = (const float*)d_in[1];
  const float* Wr[4] = {(const float*)d_in[2],  (const float*)d_in[6],
                        (const float*)d_in[10], (const float*)d_in[14]};
  const float* Wi[4] = {(const float*)d_in[3],  (const float*)d_in[7],
                        (const float*)d_in[11], (const float*)d_in[15]};
  const float* br[4] = {(const float*)d_in[4],  (const float*)d_in[8],
                        (const float*)d_in[12], (const float*)d_in[16]};
  const float* bi[4] = {(const float*)d_in[5],  (const float*)d_in[9],
                        (const float*)d_in[13], (const float*)d_in[17]};
  const float* gg[3] = {(const float*)d_in[18], (const float*)d_in[20],
                        (const float*)d_in[22]};
  const float* bb[3] = {(const float*)d_in[19], (const float*)d_in[21],
                        (const float*)d_in[23]};

  char* ws = (char*)d_ws;
  const size_t MB = 1024ull * 1024ull;
  float*    Cr = (float*)(ws + 0 * MB);
  float*    Ci = (float*)(ws + 16 * MB);
  float*    Hr = (float*)(ws + 32 * MB);
  float*    Hi = (float*)(ws + 48 * MB);
  _Float16* Fc[3] = {(_Float16*)(ws + 64 * MB),
                     (_Float16*)(ws + 80 * MB),
                     (_Float16*)(ws + 96 * MB)};

  const dim3 gGemm(8, 32);   // N/128 x M/128

  for (int p = 0; p < 3; ++p) {   // Q, K, V projections
    gemm_cplx_wmma<<<gGemm, 256, 0, stream>>>(xr, xi, 0, Wr[p], Wi[p], -1.f,
                                              br[p], Cr, 1024);
    gemm_cplx_wmma<<<gGemm, 256, 0, stream>>>(xr, xi, 0, Wi[p], Wr[p], +1.f,
                                              bi[p], Ci, 1024);
    cln_kernel<<<4096, 256, 0, stream>>>(Cr, Ci, gg[p], bb[p], Hr, Hi);
    fft1024_kernel<<<1024, 256, 0, stream>>>(Hr, Hi, -1.f, nullptr, nullptr, Fc[p]);
  }

  flash_attn_wmma<<<1024, 256, 0, stream>>>(Fc[0], Fc[1], Fc[2], Cr, Ci);

  fft1024_kernel<<<1024, 256, 0, stream>>>(Cr, Ci, +1.f, Hr, Hi, nullptr);  // ifft

  float* out = (float*)d_out;   // [B,N,2048] = [real | imag]
  gemm_cplx_wmma<<<gGemm, 256, 0, stream>>>(Hr, Hi, 1, Wr[3], Wi[3], -1.f,
                                            br[3], out, 2048);
  gemm_cplx_wmma<<<gGemm, 256, 0, stream>>>(Hr, Hi, 1, Wi[3], Wr[3], +1.f,
                                            bi[3], out + 1024, 2048);
}
